// Pytorch_DTLN_P2_stateful_49916109914572
// MI455X (gfx1250) — matrix operand order repaired and checked by reading, not compiled, not run on hardware
//
#include <hip/hip_runtime.h>
#include <hip/hip_bf16.h>
#include <stdint.h>

#define NB    64
#define TT    2048
#define FRAME 512
#define ENCC  256
#define HIDD  128
#define G4    512
#define MM    (NB * TT)   // 131072 rows of (n,t)

typedef __bf16 bf16;
typedef __attribute__((ext_vector_type(16))) __bf16 v16bf;
typedef __attribute__((ext_vector_type(8)))  __bf16 v8bf;
typedef __attribute__((ext_vector_type(8)))  float  v8f;

__device__ inline bf16 f2bf(float f) {
  union { float f; uint32_t u; } x; x.f = f;
  uint32_t r = (x.u + 0x7FFFu + ((x.u >> 16) & 1u)) >> 16;  // RNE
  uint16_t h = (uint16_t)r;
  bf16 b;
  __builtin_memcpy(&b, &h, 2);
  return b;
}

#define LOG2E 1.4426950408889634f

// Branch-free hardware transcendentals (v_exp_f32 / v_rcp_f32 / v_tanh_f32).
__device__ inline float fexp2(float x) { return __builtin_amdgcn_exp2f(x); }
__device__ inline float frcp(float x)  { return __builtin_amdgcn_rcpf(x); }
__device__ inline float sigm(float x)  { return frcp(1.0f + fexp2(x * -LOG2E)); }
__device__ inline float ftanh(float x) {
#if __has_builtin(__builtin_amdgcn_tanhf)
  return __builtin_amdgcn_tanhf(x);
#else
  float a = __builtin_fabsf(x);
  float e = fexp2(a * (-2.0f * LOG2E));          // e in (0,1], no overflow
  float t = (1.0f - e) * frcp(1.0f + e);
  return __builtin_copysignf(t, x);
#endif
}

// A fragment (16x32 bf16): per lane one contiguous 16-elem run (K = half*16).
__device__ inline v16bf ldfragA(const bf16* p) {
  v8bf a = *(const v8bf*)p;
  v8bf b = *(const v8bf*)(p + 8);
  v16bf r;
#pragma unroll
  for (int i = 0; i < 8; ++i) { r[i] = a[i]; r[i + 8] = b[i]; }
  return r;
}
// B fragment (32x16 bf16): per lane one contiguous 16-elem run (K = half*16).
__device__ inline v16bf ldfragB(const bf16* p) {
  v8bf a = *(const v8bf*)p;
  v8bf b = *(const v8bf*)(p + 8);
  v16bf r;
#pragma unroll
  for (int i = 0; i < 8; ++i) { r[i] = a[i]; r[i + 8] = b[i]; }
  return r;
}

// ---------------- small utility kernels ----------------
__global__ void k_cvt(const float* __restrict__ in, bf16* __restrict__ out, int n) {
  int i = blockIdx.x * 1024 + threadIdx.x;
#pragma unroll
  for (int j = 0; j < 4; ++j) {
    int idx = i + j * 256;
    if (idx < n) out[idx] = f2bf(in[idx]);
  }
}

__global__ void k_bias_sum(const float* __restrict__ a, const float* __restrict__ b,
                           float* __restrict__ out, int n) {
  int i = blockIdx.x * 256 + threadIdx.x;
  if (i < n) out[i] = a[i] + b[i];
}

__global__ void k_init_stats(float* stats) { stats[0] = 0.0f; stats[1] = 0.0f; }

__global__ void k_finalize(float* stats, float inv_count) {
  float mean = stats[0] * inv_count;
  float var  = stats[1] * inv_count - mean * mean;
  stats[2] = mean;
  stats[3] = rsqrtf(var + 1e-7f);
}

// y1 [N, FRAME, T] f32  ->  y1t [(n*T+t), FRAME] bf16 (transpose + convert)
__global__ __launch_bounds__(256) void k_transpose(const float* __restrict__ y1,
                                                   bf16* __restrict__ y1t) {
  __shared__ float tile[32][33];
  int tx = threadIdx.x, ty = threadIdx.y;
  int t0 = blockIdx.x * 32, c0 = blockIdx.y * 32, n = blockIdx.z;
  const float* src = y1 + (size_t)n * FRAME * TT;
#pragma unroll
  for (int i = 0; i < 4; ++i)
    tile[ty + i * 8][tx] = src[(size_t)(c0 + ty + i * 8) * TT + t0 + tx];
  __syncthreads();
  bf16* dst = y1t + (size_t)n * TT * FRAME;
#pragma unroll
  for (int i = 0; i < 4; ++i)
    dst[(size_t)(t0 + ty + i * 8) * FRAME + c0 + tx] = f2bf(tile[tx][ty + i * 8]);
}

// normed = (enc - mean) * rstd * gamma[c] + beta[c], stored bf16
__global__ void k_normed(const float* __restrict__ enc, const float* __restrict__ gamma,
                         const float* __restrict__ beta, const float* __restrict__ stats,
                         bf16* __restrict__ out, int total) {
  float mean = stats[2], rstd = stats[3];
  for (int i = blockIdx.x * blockDim.x + threadIdx.x; i < total;
       i += gridDim.x * blockDim.x) {
    int c = i & (ENCC - 1);
    out[i] = f2bf((enc[i] - mean) * rstd * gamma[c] + beta[c]);
  }
}

// ---------------- generic GEMM: C[M,NC] = A[M,K] * W[NC,K]^T, bf16 WMMA ----------------
// EPI 0: +bias, store f32.  EPI 1: sigmoid(+bias)*enc -> bf16.  EPI 2: store f32 + LN stats.
template <int EPI>
__global__ __launch_bounds__(256) void k_gemm(
    const bf16* __restrict__ A, const bf16* __restrict__ W,
    const float* __restrict__ bias, const float* __restrict__ encf,
    float* __restrict__ outF, bf16* __restrict__ outB,
    float* __restrict__ stats, int M, int K, int NC) {
  int lane = threadIdx.x & 31;
  int wave = threadIdx.x >> 5;
  int m0 = blockIdx.y * 128 + wave * 16;
  int n0 = blockIdx.x * 64;
  int mrow = lane & 15, half = lane >> 4;

  v8f acc[4];
#pragma unroll
  for (int nt = 0; nt < 4; ++nt)
#pragma unroll
    for (int r = 0; r < 8; ++r) acc[nt][r] = 0.0f;

  const bf16* Arow = A + (size_t)(m0 + mrow) * K + half * 16;
  for (int k0 = 0; k0 < K; k0 += 32) {
    if (k0 + 32 < K) __builtin_prefetch(Arow + k0 + 32, 0, 1);
    v16bf af = ldfragA(Arow + k0);
#pragma unroll
    for (int nt = 0; nt < 4; ++nt) {
      v16bf bfr = ldfragB(W + (size_t)(n0 + nt * 16 + mrow) * K + k0 + half * 16);
      acc[nt] = __builtin_amdgcn_wmma_f32_16x16x32_bf16(
          false, af, false, bfr, (short)0, acc[nt], false, false);
    }
  }

  float lsum = 0.f, lsq = 0.f;
#pragma unroll
  for (int nt = 0; nt < 4; ++nt) {
    int col = n0 + nt * 16 + mrow;
#pragma unroll
    for (int r = 0; r < 8; ++r) {
      int row = m0 + r + 8 * half;
      float v = acc[nt][r];
      if constexpr (EPI == 0) {
        outF[(size_t)row * NC + col] = v + bias[col];
      } else if constexpr (EPI == 1) {
        float s = sigm(v + bias[col]);
        outB[(size_t)row * NC + col] = f2bf(s * encf[(size_t)row * NC + col]);
      } else {
        outF[(size_t)row * NC + col] = v;
        lsum += v; lsq += v * v;
      }
    }
  }
  if constexpr (EPI == 2) {
    __shared__ float red0[256], red1[256];
    int tid = threadIdx.x;
    red0[tid] = lsum; red1[tid] = lsq;
    __syncthreads();
#pragma unroll
    for (int s = 128; s > 0; s >>= 1) {
      if (tid < s) { red0[tid] += red0[tid + s]; red1[tid] += red1[tid + s]; }
      __syncthreads();
    }
    if (tid == 0) { atomicAdd(&stats[0], red0[0]); atomicAdd(&stats[1], red1[0]); }
  }
}

// ---------------- persistent LSTM: one block per 16 batch rows ----------------
// gates[16,512] = xg[t] + h @ Whh^T ; Whh fragments live in registers for all T steps.
__global__ __launch_bounds__(256) void k_lstm(
    const float* __restrict__ xg, const bf16* __restrict__ Whh,
    const float* __restrict__ h0, const float* __restrict__ c0,
    bf16* __restrict__ xout, float* __restrict__ hT, float* __restrict__ cT) {
  __shared__ __align__(16) bf16 hsm[16 * HIDD];
  int tid = threadIdx.x, lane = tid & 31, wave = tid >> 5;
  int n0 = blockIdx.x * 16;
  int nloc = lane & 15, half = lane >> 4;
  int hcol = wave * 16 + nloc;  // wave owns hidden cols [wave*16, wave*16+16)

  // Preload Whh B-fragments (4 gates x 4 k-steps) into registers.
  v16bf Bf[4][4];
#pragma unroll
  for (int gt = 0; gt < 4; ++gt)
#pragma unroll
    for (int kk = 0; kk < 4; ++kk)
      Bf[gt][kk] = ldfragB(Whh + (size_t)(gt * HIDD + hcol) * HIDD + kk * 32 + half * 16);

  for (int i = tid; i < 16 * HIDD; i += 256)
    hsm[i] = f2bf(h0[(size_t)(n0 + (i >> 7)) * HIDD + (i & 127)]);

  float creg[8];
#pragma unroll
  for (int r = 0; r < 8; ++r)
    creg[r] = c0[(size_t)(n0 + r + 8 * half) * HIDD + hcol];

  __syncthreads();

  float hval[8];
  for (int t = 0; t < TT; ++t) {
    // gate pre-activations from precomputed input gates
    v8f acc[4];
#pragma unroll
    for (int gt = 0; gt < 4; ++gt) {
      int gcol = gt * HIDD + hcol;
#pragma unroll
      for (int r = 0; r < 8; ++r)
        acc[gt][r] = xg[((size_t)(n0 + r + 8 * half) * TT + t) * G4 + gcol];
      if (t + 1 < TT)
        __builtin_prefetch(&xg[((size_t)(n0 + 8 * half) * TT + t + 1) * G4 + gcol], 0, 1);
    }
    // recurrent matmul: h @ Whh^T
#pragma unroll
    for (int kk = 0; kk < 4; ++kk) {
      v16bf af = ldfragA(&hsm[nloc * HIDD + kk * 32 + half * 16]);
#pragma unroll
      for (int gt = 0; gt < 4; ++gt)
        acc[gt] = __builtin_amdgcn_wmma_f32_16x16x32_bf16(
            false, af, false, Bf[gt][kk], (short)0, acc[gt], false, false);
    }
    // branch-free LSTM cell update (hardware trans ops)
#pragma unroll
    for (int r = 0; r < 8; ++r) {
      float ig = sigm(acc[0][r]);
      float fg = sigm(acc[1][r]);
      float gg = ftanh(acc[2][r]);
      float og = sigm(acc[3][r]);
      float c = fg * creg[r] + ig * gg;
      creg[r] = c;
      hval[r] = og * ftanh(c);
    }
    __syncthreads();  // all waves done reading hsm
#pragma unroll
    for (int r = 0; r < 8; ++r) {
      int m = r + 8 * half;
      bf16 hb = f2bf(hval[r]);
      hsm[m * HIDD + hcol] = hb;
      xout[((size_t)(n0 + m) * TT + t) * HIDD + hcol] = hb;
    }
    __syncthreads();
  }
#pragma unroll
  for (int r = 0; r < 8; ++r) {
    int m = r + 8 * half;
    hT[(size_t)(n0 + m) * HIDD + hcol] = hval[r];
    cT[(size_t)(n0 + m) * HIDD + hcol] = creg[r];
  }
}

// ---------------- decoder: decoded[n,f,t] = sum_o W_dec[f,o] * est[n,t,o] ----------------
// A = W_dec (M=f), B = est rows (N=t) -> coalesced stores over t.
__global__ __launch_bounds__(256) void k_gemm_dec(
    const bf16* __restrict__ Wdec, const bf16* __restrict__ est,
    float* __restrict__ out) {
  int lane = threadIdx.x & 31, wave = threadIdx.x >> 5;
  int f0 = blockIdx.y * 128 + wave * 16;
  int t0 = blockIdx.x * 64;
  int nb = blockIdx.z;
  int nloc = lane & 15, half = lane >> 4;

  v8f acc[4];
#pragma unroll
  for (int nt = 0; nt < 4; ++nt)
#pragma unroll
    for (int r = 0; r < 8; ++r) acc[nt][r] = 0.0f;

  const bf16* Arow = Wdec + (size_t)(f0 + nloc) * ENCC + half * 16;
  const bf16* Bbase = est + (size_t)nb * TT * ENCC;
  for (int k0 = 0; k0 < ENCC; k0 += 32) {
    v16bf af = ldfragA(Arow + k0);
#pragma unroll
    for (int nt = 0; nt < 4; ++nt) {
      v16bf bfr = ldfragB(Bbase + (size_t)(t0 + nt * 16 + nloc) * ENCC + k0 + half * 16);
      acc[nt] = __builtin_amdgcn_wmma_f32_16x16x32_bf16(
          false, af, false, bfr, (short)0, acc[nt], false, false);
    }
  }
  float* obase = out + (size_t)nb * FRAME * TT;
#pragma unroll
  for (int nt = 0; nt < 4; ++nt)
#pragma unroll
    for (int r = 0; r < 8; ++r)
      obase[(size_t)(f0 + r + 8 * half) * TT + t0 + nt * 16 + nloc] = acc[nt][r];
}

// ---------------- host launcher ----------------
extern "C" void kernel_launch(void* const* d_in, const int* in_sizes, int n_in,
                              void* d_out, int out_size, void* d_ws, size_t ws_size,
                              hipStream_t stream) {
  (void)in_sizes; (void)n_in; (void)out_size; (void)ws_size;
  const float* y1    = (const float*)d_in[0];
  const float* st_in = (const float*)d_in[1];
  const float* Wenc  = (const float*)d_in[2];
  const float* gamma = (const float*)d_in[3];
  const float* beta  = (const float*)d_in[4];
  const float* Wih1  = (const float*)d_in[5];
  const float* Whh1  = (const float*)d_in[6];
  const float* bih1  = (const float*)d_in[7];
  const float* bhh1  = (const float*)d_in[8];
  const float* Wih2  = (const float*)d_in[9];
  const float* Whh2  = (const float*)d_in[10];
  const float* bih2  = (const float*)d_in[11];
  const float* bhh2  = (const float*)d_in[12];
  const float* Wd    = (const float*)d_in[13];
  const float* bd    = (const float*)d_in[14];
  const float* Wdec  = (const float*)d_in[15];

  float* decoded = (float*)d_out;
  float* so  = decoded + (size_t)NB * FRAME * TT;
  float* h1o = so;
  float* c1o = so + (size_t)NB * HIDD;
  float* h2o = so + (size_t)2 * NB * HIDD;
  float* c2o = so + (size_t)3 * NB * HIDD;

  char* wptr = (char*)d_ws;
  auto take = [&](size_t bytes) -> void* {
    void* p = (void*)wptr;
    wptr += (bytes + 255) & ~(size_t)255;
    return p;
  };
  float* stats = (float*)take(4 * sizeof(float));
  bf16*  y1t   = (bf16*) take((size_t)MM * FRAME * 2);
  float* enc   = (float*)take((size_t)MM * ENCC * 4);
  bf16*  normB = (bf16*) take((size_t)MM * ENCC * 2);   // reused later as est
  float* xg    = (float*)take((size_t)MM * G4 * 4);     // reused for LSTM2
  bf16*  x1    = (bf16*) take((size_t)MM * HIDD * 2);
  bf16*  x2    = (bf16*) take((size_t)MM * HIDD * 2);
  bf16*  wencB = (bf16*) take((size_t)ENCC * FRAME * 2);
  bf16*  wih1B = (bf16*) take((size_t)G4 * ENCC * 2);
  bf16*  whh1B = (bf16*) take((size_t)G4 * HIDD * 2);
  bf16*  wih2B = (bf16*) take((size_t)G4 * HIDD * 2);
  bf16*  whh2B = (bf16*) take((size_t)G4 * HIDD * 2);
  bf16*  wdB   = (bf16*) take((size_t)ENCC * HIDD * 2);
  bf16*  wdecB = (bf16*) take((size_t)FRAME * ENCC * 2);
  float* bsum1 = (float*)take((size_t)G4 * 4);
  float* bsum2 = (float*)take((size_t)G4 * 4);

  auto cvt = [&](const float* in, bf16* out, int n) {
    k_cvt<<<dim3((n + 1023) / 1024), dim3(256), 0, stream>>>(in, out, n);
  };
  cvt(Wenc, wencB, ENCC * FRAME);
  cvt(Wih1, wih1B, G4 * ENCC);
  cvt(Whh1, whh1B, G4 * HIDD);
  cvt(Wih2, wih2B, G4 * HIDD);
  cvt(Whh2, whh2B, G4 * HIDD);
  cvt(Wd,   wdB,   ENCC * HIDD);
  cvt(Wdec, wdecB, FRAME * ENCC);
  k_bias_sum<<<dim3(2), dim3(256), 0, stream>>>(bih1, bhh1, bsum1, G4);
  k_bias_sum<<<dim3(2), dim3(256), 0, stream>>>(bih2, bhh2, bsum2, G4);
  k_init_stats<<<1, 1, 0, stream>>>(stats);

  k_transpose<<<dim3(TT / 32, FRAME / 32, NB), dim3(32, 8), 0, stream>>>(y1, y1t);

  // encoder GEMM + LN partial stats
  k_gemm<2><<<dim3(ENCC / 64, MM / 128), dim3(256), 0, stream>>>(
      y1t, wencB, nullptr, nullptr, enc, nullptr, stats, MM, FRAME, ENCC);

  k_finalize<<<1, 1, 0, stream>>>(stats, 1.0f / (float)((size_t)MM * ENCC));

  k_normed<<<dim3(4096), dim3(256), 0, stream>>>(enc, gamma, beta, stats, normB,
                                                 MM * ENCC);

  // xg1 = normed @ Wih1^T + (bih1+bhh1)
  k_gemm<0><<<dim3(G4 / 64, MM / 128), dim3(256), 0, stream>>>(
      normB, wih1B, bsum1, nullptr, xg, nullptr, nullptr, MM, ENCC, G4);

  k_lstm<<<dim3(NB / 16), dim3(256), 0, stream>>>(
      xg, whh1B, st_in, st_in + (size_t)NB * HIDD, x1, h1o, c1o);

  // xg2 = x1 @ Wih2^T + (bih2+bhh2)   (reuses xg buffer)
  k_gemm<0><<<dim3(G4 / 64, MM / 128), dim3(256), 0, stream>>>(
      x1, wih2B, bsum2, nullptr, xg, nullptr, nullptr, MM, HIDD, G4);

  k_lstm<<<dim3(NB / 16), dim3(256), 0, stream>>>(
      xg, whh2B, st_in + (size_t)2 * NB * HIDD, st_in + (size_t)3 * NB * HIDD,
      x2, h2o, c2o);

  // est = sigmoid(x2 @ Wd^T + bd) * enc   (bf16, reuses normB region)
  k_gemm<1><<<dim3(ENCC / 64, MM / 128), dim3(256), 0, stream>>>(
      x2, wdB, bd, enc, nullptr, normB, nullptr, MM, HIDD, ENCC);

  // decoded = W_dec @ est (per batch), coalesced over t
  k_gemm_dec<<<dim3(TT / 64, FRAME / 128, NB), dim3(256), 0, stream>>>(
      wdecB, normB, decoded);
}
